// APPNP_1786706395679
// MI455X (gfx1250) — matrix operand order repaired and checked by reading, not compiled, not run on hardware
//
#include <hip/hip_runtime.h>
#include <hip/hip_bf16.h>
#include <stdint.h>

#define N_IN_FEATS 1024
#define N_HIDDEN   2048
#define N_CLASSES  128
#define K_ITERS    10
#define ALPHA_F    0.1f

typedef __bf16 bf16_t;
typedef __attribute__((ext_vector_type(16))) __bf16 v16bf;
typedef __attribute__((ext_vector_type(2)))  __bf16 v2bf;
typedef __attribute__((ext_vector_type(8)))  float  v8f;
typedef __attribute__((ext_vector_type(4)))  int    v4i;

#define GLOBAL_AS __attribute__((address_space(1)))
#define LDS_AS    __attribute__((address_space(3)))

typedef GLOBAL_AS v4i* g_v4i_ptr;
typedef LDS_AS    v4i* l_v4i_ptr;

#if __has_builtin(__builtin_amdgcn_global_load_async_to_lds_b128) && \
    __has_builtin(__builtin_amdgcn_s_wait_asynccnt)
#define USE_ASYNC_LDS 1
#else
#define USE_ASYNC_LDS 0
#endif

// ---------- bf16 conversion helpers (hardware cvt path) ----------
__device__ __forceinline__ uint32_t packcvt(float x, float y) {
#if __has_builtin(__builtin_amdgcn_cvt_pk_bf16_f32)
  union { v2bf v; uint32_t u; } c;
  c.v = __builtin_amdgcn_cvt_pk_bf16_f32(x, y);
  return c.u;
#else
  union { bf16_t h[2]; uint32_t u; } c;
  c.h[0] = (bf16_t)x; c.h[1] = (bf16_t)y;
  return c.u;
#endif
}

// ---------- degree / norm ----------
__global__ void deg_kernel(const int* __restrict__ src, const int* __restrict__ dst,
                           float* __restrict__ dout, float* __restrict__ din, int E) {
  int e = blockIdx.x * blockDim.x + threadIdx.x;
  if (e < E) {
    atomicAdd(&dout[src[e]], 1.0f);
    atomicAdd(&din[dst[e]], 1.0f);
  }
}
__global__ void rsqrt_kernel(float* __restrict__ d, int n) {
  int i = blockIdx.x * blockDim.x + threadIdx.x;
  if (i < n) d[i] = rsqrtf(fmaxf(d[i], 1.0f));
}

// ---------- X f32 -> bf16 (contiguous, vectorized) ----------
__global__ void f32_to_bf16_kernel(const float* __restrict__ in, bf16_t* __restrict__ out,
                                   int total4) {
  int i = blockIdx.x * blockDim.x + threadIdx.x;
  if (i < total4) {
    float4 f = ((const float4*)in)[i];
    uint2 p;
    p.x = packcvt(f.x, f.y);
    p.y = packcvt(f.z, f.w);
    ((uint2*)out)[i] = p;
  }
}

// ---------- weight conversion: W [Kd,Nd] f32 row-major -> Wt [Nd,Kd] bf16 ----------
__global__ void transpose_bf16_kernel(const float* __restrict__ W, bf16_t* __restrict__ Wt,
                                      int Kd, int Nd) {
  int idx = blockIdx.x * blockDim.x + threadIdx.x;
  if (idx < Kd * Nd) {
    int k = idx / Nd, n = idx - k * Nd;
    Wt[(size_t)n * Kd + k] = (bf16_t)W[idx];
  }
}

// ---------- WMMA GEMM: Out[M,ND] = op(A[M,KD](bf16) x Bt[ND,KD]^T + bias) ----------
// 64x64 tile / 128 threads (4 waves), each wave a 32x32 quadrant (2x2 WMMA tiles).
// K stepped by 32 through double-buffered LDS fed by async-to-LDS copies.
// KD/ND compile-time so all stride math folds into immediate offsets.
template<int KD, int ND, bool RELU, bool OUT_BF16>
__global__ __launch_bounds__(128) void gemm_wmma_kernel(
    const bf16_t* __restrict__ A, const bf16_t* __restrict__ Bt,
    const float* __restrict__ bias, void* __restrict__ Outv, int M)
{
  __shared__ __align__(16) bf16_t As[2][64][40];   // row stride 80B (16B aligned)
  __shared__ __align__(16) bf16_t Bs[2][64][40];

  const int m0   = blockIdx.x * 64;
  const int n0   = blockIdx.y * 64;
  const int tid  = threadIdx.x;
  const int wave = tid >> 5;
  const int lane = tid & 31;
  const int wm   = (wave >> 1) * 32;
  const int wn   = (wave & 1) * 32;
  const int hf   = lane >> 4;
  const int l15  = lane & 15;

  const int ldr = tid >> 1;          // 0..63: row of A-tile / row of Bt-tile
  const int ldk = (tid & 1) * 16;    // 0 or 16 within the 32-wide K slab

  const bool fullM = (m0 + 64 <= M);
  constexpr int ntiles = KD >> 5;

  v8f acc[2][2];
#pragma unroll
  for (int t = 0; t < 2; ++t)
#pragma unroll
    for (int u = 0; u < 2; ++u)
      acc[t][u] = 0.0f;

  auto load_tile = [&](int kt, int b) {
    const bf16_t* ga = A  + (size_t)(m0 + ldr) * KD + kt * 32 + ldk;
    const bf16_t* gb = Bt + (size_t)(n0 + ldr) * KD + kt * 32 + ldk;
    bf16_t* la = &As[b][ldr][ldk];
    bf16_t* lb = &Bs[b][ldr][ldk];
#if USE_ASYNC_LDS
    if (fullM) {
      g_v4i_ptr gap = (g_v4i_ptr)ga;
      l_v4i_ptr lap = (l_v4i_ptr)la;
      g_v4i_ptr gbp = (g_v4i_ptr)gb;
      l_v4i_ptr lbp = (l_v4i_ptr)lb;
      __builtin_amdgcn_global_load_async_to_lds_b128(gap, lap, 0, 0);
      __builtin_amdgcn_global_load_async_to_lds_b128(gap, lap, 16, 0);
      __builtin_amdgcn_global_load_async_to_lds_b128(gbp, lbp, 0, 0);
      __builtin_amdgcn_global_load_async_to_lds_b128(gbp, lbp, 16, 0);
      return;
    }
#endif
    // sync fallback (also handles the partial last M-block)
    uint4 p0, p1;
    if (m0 + ldr < M) {
      const uint4* g = (const uint4*)ga;
      p0 = g[0]; p1 = g[1];
      if (kt + 1 < ntiles) __builtin_prefetch(ga + 32, 0, 1);
    } else {
      p0 = make_uint4(0, 0, 0, 0);
      p1 = p0;
    }
    ((uint4*)la)[0] = p0;
    ((uint4*)la)[1] = p1;
    const uint4* g2 = (const uint4*)gb;
    ((uint4*)lb)[0] = g2[0];
    ((uint4*)lb)[1] = g2[1];
  };

  auto wait_tiles = [&]() {
#if USE_ASYNC_LDS
    if (fullM) __builtin_amdgcn_s_wait_asynccnt(0);
#endif
    __syncthreads();
  };

  load_tile(0, 0);
  wait_tiles();

  int cur = 0;
  for (int kt = 0; kt < ntiles; ++kt) {
    const bool more = (kt + 1 < ntiles);
    if (more) load_tile(kt + 1, cur ^ 1);

    // ---- fragments per documented CDNA5 VGPR layouts ----
    v16bf afrag[2], bfrag[2];
#pragma unroll
    for (int t = 0; t < 2; ++t) {
      const bf16_t* arow = &As[cur][wm + t * 16 + l15][0];
#pragma unroll
      for (int j = 0; j < 8; ++j) {
        int kk = (j < 4) ? (2 * j + 8 * hf) : (16 + 2 * (j - 4) + 8 * hf);
        union { uint32_t u; bf16_t h[2]; } x;
        x.u = *(const uint32_t*)&arow[kk];
        afrag[t][2 * j]     = x.h[0];
        afrag[t][2 * j + 1] = x.h[1];
      }
    }
#pragma unroll
    for (int u = 0; u < 2; ++u) {
      const bf16_t* brow = &Bs[cur][wn + u * 16 + l15][0];
      const int kb = hf * 8;
#pragma unroll
      for (int j = 0; j < 8; ++j) {
        union { uint32_t u32; bf16_t h[2]; } x;
        x.u32 = *(const uint32_t*)&brow[kb + 2 * j + ((j < 4) ? 0 : 8)];
        bfrag[u][2 * j]     = x.h[0];
        bfrag[u][2 * j + 1] = x.h[1];
      }
    }

#pragma unroll
    for (int t = 0; t < 2; ++t)
#pragma unroll
      for (int u = 0; u < 2; ++u)
        acc[t][u] = __builtin_amdgcn_wmma_f32_16x16x32_bf16(
            false, afrag[t], false, bfrag[u], (short)0, acc[t][u], false, false);

    if (more) {
      wait_tiles();
      cur ^= 1;
    }
  }

  // ---- epilogue: bias (+relu); full tiles take the unguarded straight-line path ----
  if (fullM) {
#pragma unroll
    for (int u = 0; u < 2; ++u) {
      int col = n0 + wn + u * 16 + l15;
      float bv = bias[col];
#pragma unroll
      for (int t = 0; t < 2; ++t) {
        size_t base = (size_t)(m0 + wm + t * 16 + 8 * hf) * ND + col;
#pragma unroll
        for (int v = 0; v < 8; ++v) {
          float x = acc[t][u][v] + bv;
          if (RELU) x = fmaxf(x, 0.0f);
          if (OUT_BF16) ((bf16_t*)Outv)[base + (size_t)v * ND] = (bf16_t)x;
          else          ((float*)Outv)[base + (size_t)v * ND] = x;
        }
      }
    }
  } else {
#pragma unroll
    for (int u = 0; u < 2; ++u) {
      int col = n0 + wn + u * 16 + l15;
      float bv = bias[col];
#pragma unroll
      for (int t = 0; t < 2; ++t) {
#pragma unroll
        for (int v = 0; v < 8; ++v) {
          int row = m0 + wm + t * 16 + v + 8 * hf;
          if (row < M) {
            float x = acc[t][u][v] + bv;
            if (RELU) x = fmaxf(x, 0.0f);
            if (OUT_BF16) ((bf16_t*)Outv)[(size_t)row * ND + col] = (bf16_t)x;
            else          ((float*)Outv)[(size_t)row * ND + col] = x;
          }
        }
      }
    }
  }
}

// ---------- propagation: one wave32 per edge, float4 per lane (128 ch) ----------
__global__ void prop_kernel(const float* __restrict__ h, const float* __restrict__ nsrc,
                            const int* __restrict__ src, const int* __restrict__ dst,
                            float* __restrict__ acc, int E) {
  int tid  = blockIdx.x * blockDim.x + threadIdx.x;
  int e    = tid >> 5;
  int lane = tid & 31;
  if (e < E) {
    int s = src[e], d = dst[e];
    float ns = nsrc[s];
    float4 v = ((const float4*)(h + (size_t)s * N_CLASSES))[lane];
    float* ap = acc + (size_t)d * N_CLASSES + lane * 4;
    atomicAdd(ap + 0, v.x * ns);
    atomicAdd(ap + 1, v.y * ns);
    atomicAdd(ap + 2, v.z * ns);
    atomicAdd(ap + 3, v.w * ns);
  }
}

// h_new = (1-alpha) * (agg * norm_dst) + alpha * h0   (in place on agg)
__global__ void combine_kernel(float* __restrict__ hn, const float* __restrict__ ndst,
                               const float* __restrict__ h0, int total) {
  int i = blockIdx.x * blockDim.x + threadIdx.x;
  if (i < total) {
    int node = i >> 7;  // /128
    hn[i] = (1.0f - ALPHA_F) * (hn[i] * ndst[node]) + ALPHA_F * h0[i];
  }
}

// ---------- host ----------
static inline size_t align256(size_t x) { return (x + 255) & ~(size_t)255; }

extern "C" void kernel_launch(void* const* d_in, const int* in_sizes, int n_in,
                              void* d_out, int out_size, void* d_ws, size_t ws_size,
                              hipStream_t stream) {
  const float* features = (const float*)d_in[0];
  const int*   edge_ix  = (const int*)d_in[1];
  const float* W1 = (const float*)d_in[2];
  const float* b1 = (const float*)d_in[3];
  const float* W2 = (const float*)d_in[4];
  const float* b2 = (const float*)d_in[5];
  (void)n_in; (void)out_size; (void)ws_size;

  const int N = in_sizes[0] / N_IN_FEATS;
  const int E = in_sizes[1] / 2;
  const int* src = edge_ix;
  const int* dst = edge_ix + E;

  char* ws = (char*)d_ws;
  size_t off = 0;
  auto take = [&](size_t bytes) -> char* {
    char* p = ws + off;
    off = align256(off + bytes);
    return p;
  };
  bf16_t* Xbf = (bf16_t*)take((size_t)N * N_IN_FEATS * sizeof(bf16_t));
  bf16_t* W1t = (bf16_t*)take((size_t)N_IN_FEATS * N_HIDDEN * sizeof(bf16_t));
  bf16_t* W2t = (bf16_t*)take((size_t)N_HIDDEN * N_CLASSES * sizeof(bf16_t));
  bf16_t* Hbf = (bf16_t*)take((size_t)N * N_HIDDEN * sizeof(bf16_t));
  float*  h0  = (float*)take((size_t)N * N_CLASSES * sizeof(float));
  float*  hA  = (float*)take((size_t)N * N_CLASSES * sizeof(float));
  float*  hB  = (float*)take((size_t)N * N_CLASSES * sizeof(float));
  float* nsrc = (float*)take((size_t)N * sizeof(float));
  float* ndst = (float*)take((size_t)N * sizeof(float));

  // degrees -> rsqrt norms
  (void)hipMemsetAsync(nsrc, 0, (size_t)N * sizeof(float), stream);
  (void)hipMemsetAsync(ndst, 0, (size_t)N * sizeof(float), stream);
  deg_kernel<<<(E + 255) / 256, 256, 0, stream>>>(src, dst, nsrc, ndst, E);
  rsqrt_kernel<<<(N + 255) / 256, 256, 0, stream>>>(nsrc, N);
  rsqrt_kernel<<<(N + 255) / 256, 256, 0, stream>>>(ndst, N);

  // features -> bf16 (hardware cvt), weights -> transposed bf16
  {
    int tot4 = (N * N_IN_FEATS) / 4;
    f32_to_bf16_kernel<<<(tot4 + 255) / 256, 256, 0, stream>>>(features, Xbf, tot4);
    int tot1 = N_IN_FEATS * N_HIDDEN;
    transpose_bf16_kernel<<<(tot1 + 255) / 256, 256, 0, stream>>>(W1, W1t, N_IN_FEATS, N_HIDDEN);
    int tot2 = N_HIDDEN * N_CLASSES;
    transpose_bf16_kernel<<<(tot2 + 255) / 256, 256, 0, stream>>>(W2, W2t, N_HIDDEN, N_CLASSES);
  }

  // GEMM1: H = relu(X W1 + b1)  (bf16 out)
  dim3 g1((N + 63) / 64, N_HIDDEN / 64);
  gemm_wmma_kernel<N_IN_FEATS, N_HIDDEN, true, true><<<g1, 128, 0, stream>>>(
      Xbf, W1t, b1, (void*)Hbf, N);

  // GEMM2: h0 = H W2 + b2  (f32 out)
  dim3 g2((N + 63) / 64, N_CLASSES / 64);
  gemm_wmma_kernel<N_HIDDEN, N_CLASSES, false, false><<<g2, 128, 0, stream>>>(
      Hbf, W2t, b2, (void*)h0, N);

  // h = h0
  (void)hipMemcpyAsync(hA, h0, (size_t)N * N_CLASSES * sizeof(float),
                       hipMemcpyDeviceToDevice, stream);

  float* cur = hA;
  float* nxt = hB;
  for (int k = 0; k < K_ITERS; ++k) {
    (void)hipMemsetAsync(nxt, 0, (size_t)N * N_CLASSES * sizeof(float), stream);
    long long threads = (long long)E * 32;
    prop_kernel<<<(unsigned)((threads + 255) / 256), 256, 0, stream>>>(
        cur, nsrc, src, dst, nxt, E);
    int tot = N * N_CLASSES;
    combine_kernel<<<(tot + 255) / 256, 256, 0, stream>>>(nxt, ndst, h0, tot);
    float* t = cur; cur = nxt; nxt = t;
  }

  (void)hipMemcpyAsync(d_out, cur, (size_t)N * N_CLASSES * sizeof(float),
                       hipMemcpyDeviceToDevice, stream);
}
